// CnnModel_70007966925195
// MI455X (gfx1250) — matrix operand order repaired and checked by reading, not compiled, not run on hardware
//
#include <hip/hip_runtime.h>
#include <hip/hip_bf16.h>
#include <stdint.h>

// ---------------------------------------------------------------------------
// CDNA5 (gfx1250) implementation of the CNN + pairwise-distance pipeline.
//   conv2/conv3 : shifted-GEMM decomposition on v_wmma_f32_16x16x32_bf16,
//                 one block per sample, activations staged to LDS via the
//                 Tensor Data Mover (tensor_load_to_lds + s_wait_tensorcnt),
//                 A-fragments held in VGPRs across all position tiles.
//   pairwise    : z@z^T on v_wmma_f32_16x16x4_f32 (full f32 precision where
//                 the |zi|^2+|zj|^2-2*dot cancellation matters).
// ---------------------------------------------------------------------------

typedef __attribute__((ext_vector_type(16))) __bf16 v16bf;
typedef __attribute__((ext_vector_type(8)))  float  v8f;
typedef __attribute__((ext_vector_type(2)))  float  v2f;
typedef __attribute__((ext_vector_type(4)))  unsigned int u32x4;
typedef __attribute__((ext_vector_type(8)))  unsigned int u32x8;
typedef __attribute__((ext_vector_type(4)))  unsigned int v4u;   // plain 16B vector

typedef __attribute__((address_space(3))) unsigned short lds_u16;
typedef __attribute__((address_space(3))) v4u            lds_v4u;

union Frag16 {
    v4u          q[2];
    unsigned int w[8];
    v16bf        v;
};

union Pack8 {
    v4u            q;
    unsigned short s[8];
};

__device__ inline unsigned short f32_to_bf16(float f) {
    union { float f; unsigned int u; } c; c.f = f;
    unsigned int u = c.u;
    if ((u & 0x7fffffffu) > 0x7f800000u)          // NaN: keep quiet NaN
        return (unsigned short)((u >> 16) | 0x0040u);
    unsigned int r = u + 0x7fffu + ((u >> 16) & 1u);  // round-to-nearest-even
    return (unsigned short)(r >> 16);
}

// A-fragment (16x32 bf16, M = row per lane, K contiguous in memory).
// Lanes 0-15: VGPR0-3 = K 0..7, VGPR4-7 = K 16..23; lanes 16-31: +8.
__device__ inline v16bf load_a_frag(const unsigned short* rowPtr, int kk, int half) {
    Frag16 f;
    f.q[0] = *(const v4u*)(rowPtr + kk + half * 8);
    f.q[1] = *(const v4u*)(rowPtr + kk + 16 + half * 8);
    return f.v;
}

// B-fragment (32x16 bf16) from LDS: N = per-lane row, K contiguous.
// Lanes 0-15: VGPR0-3 = K 0..7, VGPR4-7 = K 16..23; lanes 16-31: +8.
__device__ inline v16bf load_b_frag_lds(const lds_u16* colPtr, int kk, int half) {
    Frag16 f;
    f.q[0] = *(const lds_v4u*)(colPtr + kk + half * 8);
    f.q[1] = *(const lds_v4u*)(colPtr + kk + 16 + half * 8);
    return f.v;
}

// ---------------------------------------------------------------------------
// Tensor Data Mover: 1-D copy of n_u16 bf16 elements from global -> LDS.
// D# built per CDNA5 ISA (count=1, data_size=1 (2B),
// tile_dim0 = tensor_dim0 = n_u16, tensor_dim1 = tile_dim1 = 1, type=2).
// Wave-level op (EXEC ignored); issue from one wave, s_wait_tensorcnt, then
// block barrier.
// ---------------------------------------------------------------------------
__device__ inline void tdm_load_to_lds(unsigned lds_byte_addr,
                                       const void* gptr, unsigned n_u16) {
    unsigned long long ga = (unsigned long long)(uintptr_t)gptr;
    u32x4 g0;
    g0[0] = 1u;                                   // count=1 (valid user D#)
    g0[1] = lds_byte_addr;                        // lds_addr
    g0[2] = (unsigned)ga;                         // global_addr[31:0]
    g0[3] = (unsigned)((ga >> 32) & 0x01FFFFFFu)  // global_addr[56:32]
          | 0x80000000u;                          // type = 2 ("image")
    u32x8 g1;
    g1[0] = 0x00010000u;                          // wg_mask=0, data_size=1 (2B)
    g1[1] = (n_u16 & 0xFFFFu) << 16;              // tensor_dim0[15:0]
    g1[2] = ((n_u16 >> 16) & 0xFFFFu)             // tensor_dim0[31:16]
          | (1u << 16);                           // tensor_dim1 = 1
    g1[3] = (n_u16 & 0xFFFFu) << 16;              // tile_dim0 = n_u16
    g1[4] = 1u;                                   // tile_dim1=1, tile_dim2=0
    g1[5] = n_u16;                                // tensor_dim0_stride[31:0]
    g1[6] = 0u;                                   // stride hi, dim1_stride lo
    g1[7] = 0u;
    asm volatile("tensor_load_to_lds %0, %1" :: "s"(g0), "s"(g1) : "memory");
}

// ---------------------------------------------------------------------------
// Utility kernels
// ---------------------------------------------------------------------------
__global__ void k_zero(float* __restrict__ p, int n) {
    int i = blockIdx.x * blockDim.x + threadIdx.x;
    if (i < n) p[i] = 0.0f;
}

// conv2_w (128,64,5) f32 -> w2p[k][oc][ic] bf16
__global__ void k_pack2(const float* __restrict__ w, unsigned short* __restrict__ o) {
    int i = blockIdx.x * blockDim.x + threadIdx.x;
    if (i >= 5 * 128 * 64) return;
    int k = i / (128 * 64);
    int r = i % (128 * 64);
    int oc = r / 64, ic = r % 64;
    o[i] = f32_to_bf16(w[(oc * 64 + ic) * 5 + k]);
}

// conv3_w (256,128,3) f32 -> w3p[k][oc][ic] bf16
__global__ void k_pack3(const float* __restrict__ w, unsigned short* __restrict__ o) {
    int i = blockIdx.x * blockDim.x + threadIdx.x;
    if (i >= 3 * 256 * 128) return;
    int k = i / (256 * 128);
    int r = i % (256 * 128);
    int oc = r / 128, ic = r % 128;
    o[i] = f32_to_bf16(w[(oc * 128 + ic) * 3 + k]);
}

// ---------------------------------------------------------------------------
// conv1: 1->64 ch, k=100, pad(49,50), relu, maxpool2.  Direct f32, LDS staged.
// Output h1: (N, 256 pos, 64 ch) bf16.
// ---------------------------------------------------------------------------
__global__ __launch_bounds__(256) void k_conv1(const float* __restrict__ samples,
                                               const float* __restrict__ w1,
                                               const float* __restrict__ b1,
                                               unsigned short* __restrict__ h1) {
    __shared__ float ss[611];       // padded sample: index i <-> sample[i-49]
    __shared__ float wsm[64 * 100];
    int n = blockIdx.x;
    int tid = threadIdx.x;
    for (int i = tid; i < 6400; i += 256) wsm[i] = w1[i];
    for (int i = tid; i < 611; i += 256) {
        int x = i - 49;
        ss[i] = (x >= 0 && x < 512) ? samples[(size_t)n * 512 + x] : 0.0f;
    }
    __syncthreads();
    for (int it = 0; it < 64; ++it) {
        int idx = it * 256 + tid;     // 0..16383
        int c = idx & 63;
        int p = idx >> 6;             // pooled position 0..255
        const float* wc = wsm + c * 100;
        const float* s0 = ss + 2 * p; // ss[x+k] == sample[x+k-49]
        float a0 = 0.f, a1 = 0.f;
        for (int k = 0; k < 100; ++k) {
            float w = wc[k];
            a0 = fmaf(w, s0[k], a0);
            a1 = fmaf(w, s0[k + 1], a1);
        }
        float bb = b1[c];
        float v = fmaxf(fmaxf(a0 + bb, 0.f), fmaxf(a1 + bb, 0.f));
        h1[((size_t)n * 256 + p) * 64 + c] = f32_to_bf16(v);
    }
}

// ---------------------------------------------------------------------------
// conv2: 64->128, k=5, pad 2.  One block (8 waves) per sample.
// h1 sample slice (256x64 bf16, 32KB) staged to LDS by the TDM with a
// zero-filled 2-row halo on each side (no padding branches).
// Wave w owns oc-tile mt=w: A-fragments (5 shifts x 2 K-steps) loaded once,
// reused across all 16 position tiles.  Fused relu+maxpool2 -> h2.
// ---------------------------------------------------------------------------
__global__ __launch_bounds__(256) void k_conv2(const unsigned short* __restrict__ h1,
                                               const unsigned short* __restrict__ w2p,
                                               const float* __restrict__ b2,
                                               unsigned short* __restrict__ h2) {
    __shared__ unsigned short sm[260 * 64];   // rows -2..257, 64 ch each
    int n   = blockIdx.x;
    int tid = threadIdx.x;

    // zero the halo rows (-2,-1 and 256,257)
    if (tid < 128) sm[tid] = 0;
    else           sm[258 * 64 + (tid - 128)] = 0;
    if (tid < 32) {
        unsigned lds_off = (unsigned)(uintptr_t)(lds_u16*)&sm[128]; // row 0
        tdm_load_to_lds(lds_off, h1 + (size_t)n * 16384, 16384);
        __builtin_amdgcn_s_wait_tensorcnt(0);
    }
    __syncthreads();

    int lane = tid & 31;
    int half = lane >> 4;
    int lm   = lane & 15;
    int mt   = tid >> 5;            // wave id == oc tile 0..7
    int m    = mt * 16 + lm;
    const lds_u16* sml = (const lds_u16*)sm;

    // preload A fragments: 5 shifts x 2 K-steps, held in VGPRs
    v16bf afr[5][2];
#pragma unroll
    for (int s = 0; s < 5; ++s)
#pragma unroll
        for (int k2 = 0; k2 < 2; ++k2)
            afr[s][k2] = load_a_frag(w2p + ((size_t)s * 128 + m) * 64, k2 * 32, half);

    for (int nt = 0; nt < 16; ++nt) {
        v8f acc = {};
#pragma unroll
        for (int s = 0; s < 5; ++s) {
            // pos = nt*16+lm + s-2 ; LDS row index = pos+2 = nt*16+lm+s
            const lds_u16* bcol = sml + (nt * 16 + lm + s) * 64;
#pragma unroll
            for (int k2 = 0; k2 < 2; ++k2) {
                v16bf b = load_b_frag_lds(bcol, k2 * 32, half);
                acc = __builtin_amdgcn_wmma_f32_16x16x32_bf16(
                    false, afr[s][k2], false, b, (short)0, acc, false, false);
            }
        }
        // bias + relu + maxpool2 (pool partner = adjacent N lane)
        float pooled[8];
#pragma unroll
        for (int r = 0; r < 8; ++r) {
            float v = acc[r] + b2[mt * 16 + half * 8 + r];
            v = fmaxf(v, 0.0f);
            pooled[r] = fmaxf(v, __shfl_xor(v, 1, 32));
        }
        if ((lane & 1) == 0) {
            int pout = nt * 8 + (lm >> 1);
            int cb   = mt * 16 + half * 8;
            Pack8 pk;
#pragma unroll
            for (int r = 0; r < 8; ++r) pk.s[r] = f32_to_bf16(pooled[r]);
            *(v4u*)(h2 + ((size_t)n * 128 + pout) * 128 + cb) = pk.q;
        }
    }
}

// ---------------------------------------------------------------------------
// conv3: 128->256, k=3, pad 1.  One block (8 waves) per sample; h2 slice
// (128x128 bf16, 32KB) TDM-staged to LDS with 1-row halos.  Wave w owns
// oc tiles {2w, 2w+1}.  Fused relu + maxpool2 + positional mean
// (xor-reduce + one atomicAdd per channel) -> g (N,256) f32, pre-zeroed.
// ---------------------------------------------------------------------------
__global__ __launch_bounds__(256) void k_conv3(const unsigned short* __restrict__ h2,
                                               const unsigned short* __restrict__ w3p,
                                               const float* __restrict__ b3,
                                               float* __restrict__ g) {
    __shared__ unsigned short sm[130 * 128];  // rows -1..128, 128 ch each
    int n   = blockIdx.x;
    int tid = threadIdx.x;

    if (tid < 128) sm[tid] = 0;               // row -1
    else           sm[129 * 128 + (tid - 128)] = 0;  // row 128
    if (tid < 32) {
        unsigned lds_off = (unsigned)(uintptr_t)(lds_u16*)&sm[128]; // row 0
        tdm_load_to_lds(lds_off, h2 + (size_t)n * 16384, 16384);
        __builtin_amdgcn_s_wait_tensorcnt(0);
    }
    __syncthreads();

    int lane = tid & 31;
    int half = lane >> 4;
    int lm   = lane & 15;
    int wv   = tid >> 5;            // wave id 0..7
    const lds_u16* sml = (const lds_u16*)sm;

    for (int mtl = 0; mtl < 2; ++mtl) {
        int mt = wv * 2 + mtl;      // oc tile 0..15
        int m  = mt * 16 + lm;

        v16bf afr[3][4];
#pragma unroll
        for (int s = 0; s < 3; ++s)
#pragma unroll
            for (int k4 = 0; k4 < 4; ++k4)
                afr[s][k4] = load_a_frag(w3p + ((size_t)s * 256 + m) * 128, k4 * 32, half);

        for (int nt = 0; nt < 8; ++nt) {
            v8f acc = {};
#pragma unroll
            for (int s = 0; s < 3; ++s) {
                // pos = nt*16+lm + s-1 ; LDS row index = pos+1 = nt*16+lm+s
                const lds_u16* bcol = sml + (nt * 16 + lm + s) * 128;
#pragma unroll
                for (int k4 = 0; k4 < 4; ++k4) {
                    v16bf b = load_b_frag_lds(bcol, k4 * 32, half);
                    acc = __builtin_amdgcn_wmma_f32_16x16x32_bf16(
                        false, afr[s][k4], false, b, (short)0, acc, false, false);
                }
            }
            // bias + relu + pool, then mean contribution: the xor-sum over a
            // 16-lane half counts each pooled pair twice -> /128 == mean/64.
#pragma unroll
            for (int r = 0; r < 8; ++r) {
                int ch = mt * 16 + half * 8 + r;
                float v = acc[r] + b3[ch];
                v = fmaxf(v, 0.0f);
                float pooled = fmaxf(v, __shfl_xor(v, 1, 32));
                float s = pooled;
                s += __shfl_xor(s, 1, 32);
                s += __shfl_xor(s, 2, 32);
                s += __shfl_xor(s, 4, 32);
                s += __shfl_xor(s, 8, 32);
                if (lm == 0) atomicAdd(&g[(size_t)n * 256 + ch], s * (1.0f / 128.0f));
            }
        }
    }
}

// ---------------------------------------------------------------------------
// fc 256->128 + relu + batchnorm  ->  z (4096,128) f32
// ---------------------------------------------------------------------------
__global__ __launch_bounds__(256) void k_fc(const float* __restrict__ g,
                                            const float* __restrict__ fw,
                                            const float* __restrict__ fb,
                                            const float* __restrict__ gamma,
                                            const float* __restrict__ beta,
                                            const float* __restrict__ mean,
                                            const float* __restrict__ var,
                                            float* __restrict__ z) {
    int gid = blockIdx.x * blockDim.x + threadIdx.x;
    int n = gid >> 7, j = gid & 127;
    const float* gr = g + (size_t)n * 256;
    const float* wr = fw + (size_t)j * 256;
    float s = fb[j];
    for (int k = 0; k < 256; ++k) s = fmaf(gr[k], wr[k], s);
    s = fmaxf(s, 0.0f);
    float sc = gamma[j] * rsqrtf(var[j] + 1e-5f);
    z[gid] = sc * (s - mean[j]) + beta[j];
}

__global__ __launch_bounds__(256) void k_sq(const float* __restrict__ z,
                                            float* __restrict__ sq) {
    int n = blockIdx.x * blockDim.x + threadIdx.x;
    const float* zr = z + (size_t)n * 128;
    float s = 0.f;
    for (int k = 0; k < 128; ++k) s = fmaf(zr[k], zr[k], s);
    sq[n] = s;
}

// ---------------------------------------------------------------------------
// Pairwise: G = Z Z^T with full-precision f32 WMMA (16x16x4), then the
// contrastive-loss epilogue written straight to d_out (4096x4096 f32).
// ---------------------------------------------------------------------------
#if defined(__has_builtin)
#if __has_builtin(__builtin_amdgcn_wmma_f32_16x16x4_f32)
#define HAVE_WMMA_F32X4 1
#endif
#endif
#ifndef HAVE_WMMA_F32X4
#define HAVE_WMMA_F32X4 0
#endif

__global__ __launch_bounds__(128) void k_pair(const float* __restrict__ z,
                                              const float* __restrict__ sq,
                                              const long long* __restrict__ info,
                                              float* __restrict__ out) {
    int wave = (blockIdx.x * blockDim.x + threadIdx.x) >> 5;
    int lane = threadIdx.x & 31;
    int half = lane >> 4;
    int lm   = lane & 15;
    int It = wave >> 8;          // row tile 0..255
    int Jt = wave & 255;         // col tile 0..255

    v8f acc = {};
#if HAVE_WMMA_F32X4
    const float* za = z + (size_t)(It * 16 + lm) * 128;  // A: M rows of Z
    const float* zb = z + (size_t)(Jt * 16 + lm) * 128;  // B: Z^T cols = Z rows
#pragma unroll 8
    for (int kk = 0; kk < 128; kk += 4) {
        v2f a = *(const v2f*)(za + kk + half * 2);
        v2f b = *(const v2f*)(zb + kk + half * 2);
        acc = __builtin_amdgcn_wmma_f32_16x16x4_f32(
            false, a, false, b, (short)0, acc, false, false);
    }
#else
#pragma unroll
    for (int r = 0; r < 8; ++r) {
        const float* zi = z + (size_t)(It * 16 + half * 8 + r) * 128;
        const float* zj = z + (size_t)(Jt * 16 + lm) * 128;
        float s = 0.f;
        for (int k = 0; k < 128; ++k) s = fmaf(zi[k], zj[k], s);
        acc[r] = s;
    }
#endif

    int j = Jt * 16 + lm;
    float sqj = sq[j];
    long long wj = info[2 * j];
    long long gj = info[2 * j + 1];
#pragma unroll
    for (int r = 0; r < 8; ++r) {
        int i = It * 16 + half * 8 + r;
        float d2 = sq[i] + sqj - 2.0f * acc[r];
        float val;
        if (i == j) {
            val = 0.0f;
        } else {
            float d = sqrtf(fmaxf(d2, 0.0f));
            bool y = (info[2 * i] == wj) && (info[2 * i + 1] == 1) && (gj == 1);
            val = y ? d : fmaxf(1.0f - d, 0.0f);
        }
        out[(size_t)i * 4096 + j] = val;
    }
}

// ---------------------------------------------------------------------------
// Host launcher
// ---------------------------------------------------------------------------
extern "C" void kernel_launch(void* const* d_in, const int* in_sizes, int n_in,
                              void* d_out, int out_size, void* d_ws, size_t ws_size,
                              hipStream_t stream) {
    (void)in_sizes; (void)n_in; (void)out_size; (void)ws_size;
    const float*     samples = (const float*)d_in[0];
    const long long* info    = (const long long*)d_in[1];
    const float*     w1      = (const float*)d_in[2];
    const float*     b1      = (const float*)d_in[3];
    const float*     w2      = (const float*)d_in[4];
    const float*     b2      = (const float*)d_in[5];
    const float*     w3      = (const float*)d_in[6];
    const float*     b3      = (const float*)d_in[7];
    const float*     fw      = (const float*)d_in[8];
    const float*     fb      = (const float*)d_in[9];
    const float*     gamma   = (const float*)d_in[10];
    const float*     beta    = (const float*)d_in[11];
    const float*     mean    = (const float*)d_in[12];
    const float*     var     = (const float*)d_in[13];
    float*           out     = (float*)d_out;

    char* ws = (char*)d_ws;
    size_t off = 0;
    auto carve = [&](size_t bytes) {
        char* p = ws + off;
        off = (off + bytes + 255) & ~(size_t)255;
        return p;
    };
    unsigned short* h1  = (unsigned short*)carve((size_t)4096 * 256 * 64 * 2);   // 128 MB
    unsigned short* h2  = (unsigned short*)carve((size_t)4096 * 128 * 128 * 2);  // 128 MB
    float*          g   = (float*)carve((size_t)4096 * 256 * 4);                 // 4 MB
    float*          z   = (float*)carve((size_t)4096 * 128 * 4);                 // 2 MB
    float*          sq  = (float*)carve((size_t)4096 * 4);
    unsigned short* w2p = (unsigned short*)carve((size_t)5 * 128 * 64 * 2);
    unsigned short* w3p = (unsigned short*)carve((size_t)3 * 256 * 128 * 2);

    k_pack2<<<160,   256, 0, stream>>>(w2, w2p);
    k_pack3<<<384,   256, 0, stream>>>(w3, w3p);
    k_zero <<<4096,  256, 0, stream>>>(g, 4096 * 256);
    k_conv1<<<4096,  256, 0, stream>>>(samples, w1, b1, h1);
    k_conv2<<<4096,  256, 0, stream>>>(h1, w2p, b2, h2);
    k_conv3<<<4096,  256, 0, stream>>>(h2, w3p, b3, g);
    k_fc   <<<2048,  256, 0, stream>>>(g, fw, fb, gamma, beta, mean, var, z);
    k_sq   <<<16,    256, 0, stream>>>(z, sq);
    k_pair <<<16384, 128, 0, stream>>>(z, sq, info, out);
}
